// MaskedGroupedQuerySelfAttention_62096637166394
// MI455X (gfx1250) — compile-verified, matrix operand order checked by reading, not run on hardware
//
#include <hip/hip_runtime.h>
#include <hip/hip_bf16.h>

typedef __bf16 bf16;
typedef __attribute__((ext_vector_type(16))) __bf16 v16bf;
typedef __attribute__((ext_vector_type(8)))  __bf16 v8bf;
typedef __attribute__((ext_vector_type(8)))  float  v8f;

#define B_SZ 4
#define T_SZ 1024
#define D_SZ 2048
#define NH   16
#define NG   4
#define HD   128

// ---------- helpers ----------
__device__ __forceinline__ bf16 f2bf(float f) {
    unsigned u = __builtin_bit_cast(unsigned, f);
    unsigned r = u + 0x7FFFu + ((u >> 16) & 1u);      // RNE
    return __builtin_bit_cast(__bf16, (unsigned short)(r >> 16));
}

__device__ __forceinline__ v8f wmma_bf(v16bf a, v16bf b, v8f c) {
    return __builtin_amdgcn_wmma_f32_16x16x32_bf16(
        false, a, false, b, (short)0, c, false, false);
}

// Load one 16x32 bf16 fragment (A layout; B layout is identical with
// row = N-index when the source is stored K-contiguous per row).
// Lane L: row = base_row + (L&15); elements: K = i + 8*half (i<8),
// K = i + 8 + 8*half (i>=8), half = L>>4  -> two contiguous 16B chunks.
__device__ __forceinline__ v16bf load_frag(const bf16* __restrict__ base,
                                           int ld, int row, int kbase, int lane) {
    int m = lane & 15, h = lane >> 4;
    const bf16* p = base + (size_t)(row + m) * ld + kbase + 8 * h;
    v8bf lo = *(const v8bf*)p;
    v8bf hi = *(const v8bf*)(p + 16);
    return __builtin_shufflevector(lo, hi, 0,1,2,3,4,5,6,7,8,9,10,11,12,13,14,15);
}

// ---------- elementwise conversion ----------
__global__ void convert_bf16_kernel(const float* __restrict__ in,
                                    bf16* __restrict__ out, int n) {
    int i = blockIdx.x * 256 + threadIdx.x;
    if (i < n) out[i] = f2bf(in[i]);
}

// in [rows, cols] f32  ->  out [cols, rows] bf16
__global__ void transpose_bf16_kernel(const float* __restrict__ in,
                                      bf16* __restrict__ out, int rows, int cols) {
    int i = blockIdx.x * 256 + threadIdx.x;
    if (i < rows * cols) {
        int r = i / cols, c = i - r * cols;
        out[(size_t)c * rows + r] = f2bf(in[i]);
    }
}

// ---------- GEMM: C[M,N] = A[M,K] * Bt[N,K]^T  (bf16 in, f32 acc) ----------
// One wave computes a 64x64 tile: 4x4 subtiles of 16x16, B-fragments reused
// across all 4 M-subtiles -> ~32 flop/byte of fragment traffic.
template <bool OUT_F32>
__global__ __launch_bounds__(32)
void gemm_bf16_kernel(const bf16* __restrict__ A, const bf16* __restrict__ Bt,
                      void* __restrict__ Cout, int M, int N, int K) {
    const int lane = threadIdx.x & 31;
    const int m0 = blockIdx.y * 64;
    const int n0 = blockIdx.x * 64;

    v8f acc[4][4];
    #pragma unroll
    for (int mt = 0; mt < 4; ++mt)
        #pragma unroll
        for (int nt = 0; nt < 4; ++nt)
            #pragma unroll
            for (int r = 0; r < 8; ++r) acc[mt][nt][r] = 0.0f;

    for (int kc = 0; kc < K; kc += 32) {
        v16bf bfr[4];
        #pragma unroll
        for (int nt = 0; nt < 4; ++nt)
            bfr[nt] = load_frag(Bt, K, n0 + 16 * nt, kc, lane);
        #pragma unroll
        for (int mt = 0; mt < 4; ++mt) {
            v16bf a = load_frag(A, K, m0 + 16 * mt, kc, lane);
            #pragma unroll
            for (int nt = 0; nt < 4; ++nt)
                acc[mt][nt] = wmma_bf(a, bfr[nt], acc[mt][nt]);
        }
    }

    const int n = lane & 15, h = lane >> 4;
    #pragma unroll
    for (int mt = 0; mt < 4; ++mt)
        #pragma unroll
        for (int nt = 0; nt < 4; ++nt)
            #pragma unroll
            for (int r = 0; r < 8; ++r) {
                size_t row = (size_t)(m0 + mt * 16 + r + 8 * h);
                size_t col = (size_t)(n0 + nt * 16 + n);
                float v = acc[mt][nt][r];
                if constexpr (OUT_F32) ((float*)Cout)[row * N + col] = v;
                else                   ((bf16*)Cout)[row * N + col] = f2bf(v);
            }
}

// ---------- RoPE: raw [B,T,nh,128] -> out [B,nh,T,128] ----------
__global__ void rope_kernel(const bf16* __restrict__ raw,
                            bf16* __restrict__ out, int nh) {
    int idx = blockIdx.x * 256 + threadIdx.x;
    int total = B_SZ * T_SZ * nh * 64;
    if (idx >= total) return;
    int i = idx & 63;
    int tmp = idx >> 6;
    int hh = tmp % nh; tmp /= nh;
    int t = tmp & (T_SZ - 1);
    int b = tmp >> 10;
    const bf16* p = raw + (((size_t)(b * T_SZ + t) * nh + hh) << 7);
    float q0 = (float)p[i], q1 = (float)p[i + 64];
    float inv = __expf((float)i * (-2.0f / 128.0f) * 9.210340371976184f); // ln 1e4
    float ang = (float)t * inv;
    float s, c;
    __sincosf(ang, &s, &c);
    bf16* o = out + (((size_t)(b * nh + hh) * T_SZ + t) << 7);
    o[i]      = f2bf(q0 * c - q1 * s);
    o[i + 64] = f2bf(q1 * c + q0 * s);
}

// ---------- pack V: raw [B,T,g,128] -> vt [B,g,128,T] ----------
__global__ void pack_v_kernel(const bf16* __restrict__ raw, bf16* __restrict__ vt) {
    int idx = blockIdx.x * 256 + threadIdx.x;
    if (idx >= B_SZ * T_SZ * NG * HD) return;
    int d = idx & 127;
    int tmp = idx >> 7;
    int g = tmp & 3; tmp >>= 2;
    int t = tmp & (T_SZ - 1);
    int b = tmp >> 10;
    vt[(((size_t)(b * NG + g) * HD + d) << 10) + t] = raw[idx];
}

// ---------- flash attention: one wave per (b, h, 16-row q tile) ----------
__global__ __launch_bounds__(32)
void attn_kernel(const bf16* __restrict__ Qbf, const bf16* __restrict__ Kbf,
                 const bf16* __restrict__ Vt, bf16* __restrict__ Ybf) {
    __shared__ __align__(16) bf16 ldsP[16 * 32];   // 16 q rows x 32 keys

    const int lane = threadIdx.x;
    const int qt = blockIdx.x;                      // q tile (0..63)
    const int hh = blockIdx.y;                      // head
    const int b  = blockIdx.z;
    const int g  = hh >> 2;                         // kv head (GQA 4:1)
    const int q0 = qt * 16;
    const int n  = lane & 15, hf = lane >> 4;
    const float scale = 0.08838834764831845f;       // 1/sqrt(128)

    const bf16* Qp = Qbf + (((size_t)(b * NH + hh) * T_SZ) << 7);
    const bf16* Kp = Kbf + (((size_t)(b * NG + g) * T_SZ) << 7);
    const bf16* Vp = Vt  + (((size_t)(b * NG + g) * HD) << 10);

    v16bf aq[4];
    #pragma unroll
    for (int kc = 0; kc < 4; ++kc) aq[kc] = load_frag(Qp, HD, q0, kc * 32, lane);

    v8f o[8];
    float m[8], l[8];
    #pragma unroll
    for (int ft = 0; ft < 8; ++ft)
        #pragma unroll
        for (int r = 0; r < 8; ++r) o[ft][r] = 0.0f;
    #pragma unroll
    for (int r = 0; r < 8; ++r) { m[r] = -__builtin_inff(); l[r] = 0.0f; }

    for (int sb = 0; sb <= q0 + 15; sb += 32) {
        const bool t1v = (sb + 16) <= (q0 + 15);
        v8f s0, s1;
        #pragma unroll
        for (int r = 0; r < 8; ++r) { s0[r] = 0.0f; s1[r] = 0.0f; }
        #pragma unroll
        for (int kc = 0; kc < 4; ++kc) {
            v16bf bk0 = load_frag(Kp, HD, sb, kc * 32, lane);
            s0 = wmma_bf(aq[kc], bk0, s0);
            if (t1v) {
                v16bf bk1 = load_frag(Kp, HD, sb + 16, kc * 32, lane);
                s1 = wmma_bf(aq[kc], bk1, s1);
            }
        }
        // scale + causal mask + online softmax (rows r+8*hf, cols = lane&15)
        float p0[8], p1[8], mx[8];
        #pragma unroll
        for (int r = 0; r < 8; ++r) {
            int qrow = q0 + r + 8 * hf;
            float v0 = (sb + n <= qrow) ? s0[r] * scale : -__builtin_inff();
            float v1 = (t1v && (sb + 16 + n) <= qrow) ? s1[r] * scale
                                                      : -__builtin_inff();
            p0[r] = v0; p1[r] = v1; mx[r] = fmaxf(v0, v1);
        }
        #pragma unroll
        for (int off = 1; off < 16; off <<= 1)
            #pragma unroll
            for (int r = 0; r < 8; ++r)
                mx[r] = fmaxf(mx[r], __shfl_xor(mx[r], off, 32));
        float fac[8], rs[8];
        #pragma unroll
        for (int r = 0; r < 8; ++r) {
            float mn = fmaxf(m[r], mx[r]);
            fac[r] = __expf(m[r] - mn);
            m[r] = mn;
            p0[r] = __expf(p0[r] - mn);
            p1[r] = __expf(p1[r] - mn);
            rs[r] = p0[r] + p1[r];
        }
        #pragma unroll
        for (int off = 1; off < 16; off <<= 1)
            #pragma unroll
            for (int r = 0; r < 8; ++r)
                rs[r] += __shfl_xor(rs[r], off, 32);
        #pragma unroll
        for (int r = 0; r < 8; ++r) l[r] = l[r] * fac[r] + rs[r];
        #pragma unroll
        for (int ft = 0; ft < 8; ++ft)
            #pragma unroll
            for (int r = 0; r < 8; ++r) o[ft][r] *= fac[r];

        // D-layout -> A-layout transpose of P via LDS (in-wave)
        #pragma unroll
        for (int r = 0; r < 8; ++r) {
            int row = r + 8 * hf;
            ldsP[row * 32 + n]      = f2bf(p0[r]);
            ldsP[row * 32 + 16 + n] = f2bf(p1[r]);
        }
        asm volatile("s_wait_dscnt 0x0" ::: "memory");
        v16bf pa;
        {
            const bf16* pp = ldsP + (lane & 15) * 32 + 8 * hf;
            v8bf lo = *(const v8bf*)pp;
            v8bf hi = *(const v8bf*)(pp + 16);
            pa = __builtin_shufflevector(lo, hi,
                 0,1,2,3,4,5,6,7,8,9,10,11,12,13,14,15);
        }
        // P @ V : B-frag from Vt rows = features, K-dim = keys (contiguous)
        #pragma unroll
        for (int ft = 0; ft < 8; ++ft) {
            v16bf bv = load_frag(Vp, T_SZ, ft * 16, sb, lane);
            o[ft] = wmma_bf(pa, bv, o[ft]);
        }
    }

    // epilogue: normalize and store bf16 to Y [B,T,D], D index = hh*128+ft*16+n
    #pragma unroll
    for (int ft = 0; ft < 8; ++ft)
        #pragma unroll
        for (int r = 0; r < 8; ++r) {
            int t = q0 + r + 8 * hf;
            float val = o[ft][r] / l[r];
            Ybf[(((size_t)(b * T_SZ + t)) << 11) + hh * HD + ft * 16 + n] = f2bf(val);
        }
}

// ---------- host ----------
extern "C" void kernel_launch(void* const* d_in, const int* in_sizes, int n_in,
                              void* d_out, int out_size, void* d_ws, size_t ws_size,
                              hipStream_t stream) {
    const float* x  = (const float*)d_in[0];
    const float* Wq = (const float*)d_in[1];
    const float* Wk = (const float*)d_in[2];
    const float* Wv = (const float*)d_in[3];
    const float* Wo = (const float*)d_in[4];

    const int M = B_SZ * T_SZ;          // 4096
    char* ws = (char*)d_ws;
    size_t off = 0;
    auto alloc = [&](size_t bytes) -> void* {
        void* p = ws + off;
        off += (bytes + 255) & ~(size_t)255;
        return p;
    };
    bf16* xbf  = (bf16*)alloc((size_t)M * D_SZ * 2);
    bf16* WqT  = (bf16*)alloc((size_t)D_SZ * D_SZ * 2);
    bf16* WkT  = (bf16*)alloc((size_t)512 * D_SZ * 2);
    bf16* WvT  = (bf16*)alloc((size_t)512 * D_SZ * 2);
    bf16* WoT  = (bf16*)alloc((size_t)D_SZ * D_SZ * 2);
    bf16* Qraw = (bf16*)alloc((size_t)M * D_SZ * 2);
    bf16* Kraw = (bf16*)alloc((size_t)M * 512 * 2);
    bf16* Vraw = (bf16*)alloc((size_t)M * 512 * 2);
    bf16* Qbf  = (bf16*)alloc((size_t)M * D_SZ * 2);
    bf16* Kbf  = (bf16*)alloc((size_t)M * 512 * 2);
    bf16* Vt   = (bf16*)alloc((size_t)M * 512 * 2);
    bf16* Ybf  = Qraw;                   // reuse: Qraw consumed by rope first

    // stage 0: precision conversion / weight transpose
    convert_bf16_kernel<<<(M * D_SZ + 255) / 256, 256, 0, stream>>>(x, xbf, M * D_SZ);
    transpose_bf16_kernel<<<(D_SZ * D_SZ + 255) / 256, 256, 0, stream>>>(Wq, WqT, D_SZ, D_SZ);
    transpose_bf16_kernel<<<(D_SZ * 512 + 255) / 256, 256, 0, stream>>>(Wk, WkT, D_SZ, 512);
    transpose_bf16_kernel<<<(D_SZ * 512 + 255) / 256, 256, 0, stream>>>(Wv, WvT, D_SZ, 512);
    transpose_bf16_kernel<<<(D_SZ * D_SZ + 255) / 256, 256, 0, stream>>>(Wo, WoT, D_SZ, D_SZ);

    // stage 1: QKV projections (WMMA)
    gemm_bf16_kernel<false><<<dim3(D_SZ / 64, M / 64), 32, 0, stream>>>(xbf, WqT, Qraw, M, D_SZ, D_SZ);
    gemm_bf16_kernel<false><<<dim3(512 / 64,  M / 64), 32, 0, stream>>>(xbf, WkT, Kraw, M, 512, D_SZ);
    gemm_bf16_kernel<false><<<dim3(512 / 64,  M / 64), 32, 0, stream>>>(xbf, WvT, Vraw, M, 512, D_SZ);

    // stage 2: RoPE + layout packs
    rope_kernel<<<(M * NH * 64 + 255) / 256, 256, 0, stream>>>(Qraw, Qbf, NH);
    rope_kernel<<<(M * NG * 64 + 255) / 256, 256, 0, stream>>>(Kraw, Kbf, NG);
    pack_v_kernel<<<(M * 512 + 255) / 256, 256, 0, stream>>>(Vraw, Vt);

    // stage 3: causal flash attention (WMMA)
    attn_kernel<<<dim3(T_SZ / 16, NH, B_SZ), 32, 0, stream>>>(Qbf, Kbf, Vt, Ybf);

    // stage 4: output projection (WMMA, f32 out)
    gemm_bf16_kernel<true><<<dim3(D_SZ / 64, M / 64), 32, 0, stream>>>(Ybf, WoT, d_out, M, D_SZ, D_SZ);
}
